// FeaturesMemory_41274635714665
// MI455X (gfx1250) — compile-verified, matrix operand order checked by reading, not run on hardware
//
#include <hip/hip_runtime.h>
#include <stdint.h>
#include <stddef.h>

// ---------------- types / constants ----------------
typedef __attribute__((ext_vector_type(16))) __bf16 v16bf;
typedef __attribute__((ext_vector_type(8)))  float  v8f;

#define Bb   8192        // batch
#define Dd   64          // feat channels
#define HWp  16          // h*w
#define Mm   (Bb*HWp)    // 131072 pixel-rows
#define Tt   256         // transform channels
#define CFc  1024        // feats_channels
#define Kk   4           // memory slots
#define Cc   16          // classes
#define Oo   512         // bottleneck out channels
#define CIN  1088        // D + CF

static __device__ __forceinline__ unsigned short f2bf(float f) {
  union { float f; uint32_t u; } x; x.f = f;
  uint32_t r = x.u + 0x7FFFu + ((x.u >> 16) & 1u);   // round-to-nearest-even
  return (unsigned short)(r >> 16);
}

union Frag { v16bf v; uint32_t u[8]; };

// K offset of VGPR-pair v for this lane-half (CDNA5 16-bit A/B 16x32 layout)
static __device__ __forceinline__ int koff(int v, int half) {
  return ((v & 4) << 2) + half * 8 + ((v & 3) << 1);
}

static __device__ __forceinline__ v8f wmma_bf16(v16bf a, v16bf b, v8f c) {
  return __builtin_amdgcn_wmma_f32_16x16x32_bf16(false, a, false, b, (short)0, c,
                                                 false, false);
}

// ---------------- elementwise / glue kernels ----------------

__global__ void k_cvt(const float* __restrict__ S, unsigned short* __restrict__ D, int n) {
  int i = blockIdx.x * blockDim.x + threadIdx.x;
  if (i < n) D[i] = f2bf(S[i]);
}

// bott_w (O,CIN,3,3) -> (tap, O, CIN) bf16
__global__ void k_cvt_bott(const float* __restrict__ S, unsigned short* __restrict__ D) {
  int i = blockIdx.x * blockDim.x + threadIdx.x;
  if (i >= Oo * CIN * 9) return;
  int t = i % 9;
  int c = (i / 9) % CIN;
  int o = i / (9 * CIN);
  D[((size_t)t * Oo + o) * CIN + c] = f2bf(S[i]);
}

// feats (B,D,4,4) -> Xff[b, c*16+p] (flat bf16) and Xpm[(b*16+p), c] (pixel-major)
__global__ void k_prep(const float* __restrict__ F, unsigned short* __restrict__ Xff,
                       unsigned short* __restrict__ Xpm) {
  size_t i = (size_t)blockIdx.x * blockDim.x + threadIdx.x;
  if (i >= (size_t)Bb * Dd * HWp) return;
  int p = (int)(i & 15);
  int c = (int)((i >> 4) & 63);
  size_t b = i >> 10;
  unsigned short h = f2bf(F[i]);
  Xff[i] = h;
  Xpm[(b * 16 + p) * 64 + c] = h;
}

// softmax over classes: scores[b, c*4+k] -> SM[b, k*16+c]
__global__ void k_smx(const float* __restrict__ S, float* __restrict__ SM) {
  int i = blockIdx.x * blockDim.x + threadIdx.x;
  if (i >= Bb * Kk) return;
  int b = i >> 2, k = i & 3;
  const float* row = S + (size_t)b * 64;
  float mx = -1e30f;
  for (int c = 0; c < 16; ++c) mx = fmaxf(mx, row[c * 4 + k]);
  float e[16], sum = 0.f;
  for (int c = 0; c < 16; ++c) { e[c] = __expf(row[c * 4 + k] - mx); sum += e[c]; }
  float inv = 1.f / sum;
  for (int c = 0; c < 16; ++c) SM[(size_t)b * 64 + k * 16 + c] = e[c] * inv;
}

// sel: Xk[k][(b*16+p), c] = sum_cls SM[b,k,cls] * mem[cls,k, c*16+p]  (bf16)
__global__ void k_sel(const float* __restrict__ SM, const float* __restrict__ Mem,
                      unsigned short* __restrict__ Xk) {
  size_t i = (size_t)blockIdx.x * blockDim.x + threadIdx.x;
  if (i >= (size_t)Kk * Mm * 64) return;
  int c = (int)(i & 63);
  size_t m = (i >> 6) & (Mm - 1);
  int k = (int)(i >> 23);                 // Mm*64 == 2^23
  int b = (int)(m >> 4);
  int p = (int)(m & 15);
  const float* smr = SM + (size_t)b * 64 + k * 16;
  const float* mr  = Mem + (size_t)k * 1024 + c * 16 + p;
  float s = 0.f;
#pragma unroll
  for (int cls = 0; cls < 16; ++cls) s += smr[cls] * mr[(size_t)cls * 4096];
  Xk[i] = f2bf(s);
}

// fuse epilogue: relu(fs*acc+fb) -> xc channels [64, 1088)
__global__ void k_fuse_ep(const float* __restrict__ Acc, const float* __restrict__ fs,
                          const float* __restrict__ fb, unsigned short* __restrict__ Xc) {
  size_t i = (size_t)blockIdx.x * blockDim.x + threadIdx.x;
  if (i >= (size_t)Mm * CFc) return;
  int o = (int)(i & 1023);
  size_t m = i >> 10;
  float v = fs[o] * Acc[i] + fb[o];
  v = v > 0.f ? v : 0.f;
  Xc[m * CIN + 64 + o] = f2bf(v);
}

// copy feats channels into xc channels [0, 64)
__global__ void k_xcf(const unsigned short* __restrict__ Xpm, unsigned short* __restrict__ Xc) {
  size_t i = (size_t)blockIdx.x * blockDim.x + threadIdx.x;
  if (i >= (size_t)Mm * 64) return;
  int c = (int)(i & 63);
  size_t m = i >> 6;
  Xc[m * CIN + c] = Xpm[i];
}

__global__ void k_copymem(const float* __restrict__ Mem, float* __restrict__ O) {
  int i = blockIdx.x * blockDim.x + threadIdx.x;
  if (i < Cc * Kk * CFc) O[i] = Mem[i];
}

// ---------------- 4x4-blocked WMMA GEMM: D[m,n] = sum_k A[m,k]*W[n,k] ----------------
// Each wave computes a 64x64 tile: 16 WMMA per K-step, 8 fragment loads.
// B-frags loaded once per K-step; A-frags streamed one at a time to cap liveness.
// launch_bounds(256,1) lifts the VGPR cap so the 16 f32 accumulators stay resident.
// mode 0: store f32   mode 1: f32 +=   mode 2: bf16 relu(s*x+b)
// Requires M % 64 == 0, N % 64 == 0, Kd % 32 == 0.
__global__ void __launch_bounds__(256, 1)
k_gemm4(const unsigned short* __restrict__ A, int lda,
        const unsigned short* __restrict__ W, int ldw,
        float* __restrict__ OutF, unsigned short* __restrict__ OutH,
        int ldo, const float* __restrict__ sc, const float* __restrict__ bi,
        int M, int N, int Kd, int mode) {
  int lane = threadIdx.x & 31;
  int wid  = blockIdx.x * (blockDim.x >> 5) + (threadIdx.x >> 5);
  int ntb = N >> 6, mtb = M >> 6;
  if (wid >= ntb * mtb) return;
  int mt = wid / ntb, nt = wid - mt * ntb;
  int half = lane >> 4, r16 = lane & 15;
  const unsigned short* Ar[4];
  const unsigned short* Wr[4];
#pragma unroll
  for (int i = 0; i < 4; ++i) {
    Ar[i] = A + (size_t)(mt * 64 + i * 16 + r16) * lda;
    Wr[i] = W + (size_t)(nt * 64 + i * 16 + r16) * ldw;
  }
  v8f acc[4][4] = {};
  for (int kb = 0; kb < Kd; kb += 32) {
    Frag b[4];
#pragma unroll
    for (int j = 0; j < 4; ++j)
#pragma unroll
      for (int v = 0; v < 8; ++v)
        b[j].u[v] = *(const uint32_t*)(Wr[j] + kb + koff(v, half));
    if (kb + 32 < Kd) {
      __builtin_prefetch(Ar[0] + kb + 32, 0, 1);
      __builtin_prefetch(Wr[0] + kb + 32, 0, 1);
    }
#pragma unroll
    for (int i = 0; i < 4; ++i) {
      Frag a;
#pragma unroll
      for (int v = 0; v < 8; ++v)
        a.u[v] = *(const uint32_t*)(Ar[i] + kb + koff(v, half));
#pragma unroll
      for (int j = 0; j < 4; ++j)
        acc[i][j] = wmma_bf16(a.v, b[j].v, acc[i][j]);
    }
  }
#pragma unroll
  for (int j = 0; j < 4; ++j) {
    int n = nt * 64 + j * 16 + r16;
    if (mode == 2) {
      float s = sc[n], bbv = bi[n];
#pragma unroll
      for (int i = 0; i < 4; ++i) {
        int row0 = mt * 64 + i * 16 + half * 8;
#pragma unroll
        for (int r = 0; r < 8; ++r) {
          float v = s * acc[i][j][r] + bbv;
          v = v > 0.f ? v : 0.f;
          OutH[(size_t)(row0 + r) * ldo + n] = f2bf(v);
        }
      }
    } else if (mode == 1) {
#pragma unroll
      for (int i = 0; i < 4; ++i) {
        int row0 = mt * 64 + i * 16 + half * 8;
#pragma unroll
        for (int r = 0; r < 8; ++r) {
          size_t idx = (size_t)(row0 + r) * ldo + n;
          OutF[idx] += acc[i][j][r];
        }
      }
    } else {
#pragma unroll
      for (int i = 0; i < 4; ++i) {
        int row0 = mt * 64 + i * 16 + half * 8;
#pragma unroll
        for (int r = 0; r < 8; ++r)
          OutF[(size_t)(row0 + r) * ldo + n] = acc[i][j][r];
      }
    }
  }
}

// ---------------- sim: per-batch 16x16 = Q(16x256) . K^T, scale, softmax ----------------
__global__ void k_sim(const unsigned short* __restrict__ Q,
                      const unsigned short* __restrict__ Kb,
                      unsigned short* __restrict__ P) {
  int lane = threadIdx.x & 31;
  int b = blockIdx.x * (blockDim.x >> 5) + (threadIdx.x >> 5);
  if (b >= Bb) return;
  int half = lane >> 4, r16 = lane & 15;
  const unsigned short* Ar = Q  + (size_t)(b * 16 + r16) * Tt;
  const unsigned short* Wr = Kb + (size_t)(b * 16 + r16) * Tt;
  v8f acc = {};
  for (int kb = 0; kb < Tt; kb += 32) {
    Frag a, w;
#pragma unroll
    for (int v = 0; v < 8; ++v) {
      int kp = kb + koff(v, half);
      a.u[v] = *(const uint32_t*)(Ar + kp);
      w.u[v] = *(const uint32_t*)(Wr + kp);
    }
    acc = wmma_bf16(a.v, w.v, acc);
  }
  // rows of D = query pixel, cols = key pixel; softmax over cols (across 16-lane half)
#pragma unroll
  for (int r = 0; r < 8; ++r) {
    float v = acc[r] * 0.0625f;                 // 256^-0.5
    float mx = v;
    for (int msk = 8; msk >= 1; msk >>= 1) mx = fmaxf(mx, __shfl_xor(mx, msk, 32));
    float e = __expf(v - mx);
    float sum = e;
    for (int msk = 8; msk >= 1; msk >>= 1) sum += __shfl_xor(sum, msk, 32);
    P[(size_t)(b * 16 + half * 8 + r) * 16 + r16] = f2bf(e / sum);
  }
}

// ---------------- ctx: per (batch, t-tile): P(16x16) . V(16 x 256-chunk) ----------------
__global__ void k_ctx(const unsigned short* __restrict__ P,
                      const unsigned short* __restrict__ V,
                      unsigned short* __restrict__ Ctx) {
  int lane = threadIdx.x & 31;
  int wid = blockIdx.x * (blockDim.x >> 5) + (threadIdx.x >> 5);
  if (wid >= Bb * (Tt / 16)) return;
  int b = wid >> 4, tt = wid & 15;
  int half = lane >> 4, r16 = lane & 15;
  const unsigned short* Ar = P + (size_t)(b * 16 + r16) * 16;
  const unsigned short* Vb = V + (size_t)b * 16 * Tt + tt * 16 + r16;  // B[k,n]=V[k*Tt+n]
  Frag a, w;
#pragma unroll
  for (int v = 0; v < 8; ++v) {
    int kp = koff(v, half);
    a.u[v] = (kp < 16) ? *(const uint32_t*)(Ar + kp) : 0u;
    uint32_t lo = (kp     < 16) ? (uint32_t)Vb[(size_t)kp       * Tt] : 0u;
    uint32_t hi = (kp + 1 < 16) ? (uint32_t)Vb[(size_t)(kp + 1) * Tt] : 0u;
    w.u[v] = lo | (hi << 16);
  }
  v8f acc = {};
  acc = wmma_bf16(a.v, w.v, acc);
#pragma unroll
  for (int r = 0; r < 8; ++r)
    Ctx[(size_t)(b * 16 + half * 8 + r) * Tt + tt * 16 + r16] = f2bf(acc[r]);
}

// ---------------- bott 3x3 conv: 9 tap-shifted GEMMs, 4 batches x 64 out-chans/wave ---
__global__ void __launch_bounds__(256, 1)
k_bott(const unsigned short* __restrict__ X,   // xc_pm M x CIN
       const unsigned short* __restrict__ Wt,  // (9, 512, 1088)
       const float* __restrict__ bs, const float* __restrict__ bb,
       float* __restrict__ Y) {                 // (B, 512, 16)
  int lane = threadIdx.x & 31;
  int wid = blockIdx.x * (blockDim.x >> 5) + (threadIdx.x >> 5);
  if (wid >= (Bb / 4) * (Oo / 64)) return;
  int bg = wid >> 3;              // group of 4 batches (Oo/64 == 8)
  int ot = wid & 7;               // 64-out-channel block
  int b0 = bg * 4;
  int half = lane >> 4, r16 = lane & 15;   // r16 = output pixel (A rows) / out-chan col
  int h = r16 >> 2, w2 = r16 & 3;
  const unsigned short* Wr0[4];
#pragma unroll
  for (int j = 0; j < 4; ++j)
    Wr0[j] = Wt + (size_t)(ot * 64 + j * 16 + r16) * CIN;
  v8f acc[4][4] = {};
  for (int tap = 0; tap < 9; ++tap) {
    int dy = tap / 3 - 1, dx = tap % 3 - 1;
    int hi2 = h + dy, wi = w2 + dx;
    bool valid = (hi2 >= 0) & (hi2 < 4) & (wi >= 0) & (wi < 4);
    int sp = valid ? (hi2 * 4 + wi) : 0;
    const unsigned short* Ar[4];
#pragma unroll
    for (int i = 0; i < 4; ++i)
      Ar[i] = X + (size_t)((b0 + i) * 16 + sp) * CIN;
    size_t wofs = (size_t)tap * Oo * CIN;
    for (int kb = 0; kb < CIN; kb += 32) {
      Frag wv[4];
#pragma unroll
      for (int j = 0; j < 4; ++j)
#pragma unroll
        for (int v = 0; v < 8; ++v)
          wv[j].u[v] = *(const uint32_t*)(Wr0[j] + wofs + kb + koff(v, half));
#pragma unroll
      for (int i = 0; i < 4; ++i) {
        Frag a;
#pragma unroll
        for (int v = 0; v < 8; ++v)
          a.u[v] = valid ? *(const uint32_t*)(Ar[i] + kb + koff(v, half)) : 0u;
#pragma unroll
        for (int j = 0; j < 4; ++j)
          acc[i][j] = wmma_bf16(a.v, wv[j].v, acc[i][j]);
      }
    }
  }
#pragma unroll
  for (int j = 0; j < 4; ++j) {
    int o = ot * 64 + j * 16 + r16;
    float s = bs[o], bi2 = bb[o];
#pragma unroll
    for (int i = 0; i < 4; ++i) {
#pragma unroll
      for (int r = 0; r < 8; ++r) {
        float v = s * acc[i][j][r] + bi2;
        v = v > 0.f ? v : 0.f;
        Y[(size_t)(b0 + i) * (Oo * 16) + (size_t)o * 16 + half * 8 + r] = v;
      }
    }
  }
}

// ---------------- host ----------------
extern "C" void kernel_launch(void* const* d_in, const int* in_sizes, int n_in,
                              void* d_out, int out_size, void* d_ws, size_t ws_size,
                              hipStream_t stream) {
  const float* feats  = (const float*)d_in[0];
  const float* memory = (const float*)d_in[1];
  const float* q1_w = (const float*)d_in[2];
  const float* q1_s = (const float*)d_in[3];
  const float* q1_b = (const float*)d_in[4];
  const float* k1_w = (const float*)d_in[5];
  const float* k1_s = (const float*)d_in[6];
  const float* k1_b = (const float*)d_in[7];
  const float* v_w  = (const float*)d_in[8];
  const float* v_s  = (const float*)d_in[9];
  const float* v_b  = (const float*)d_in[10];
  const float* q2_w = (const float*)d_in[11];
  const float* q2_s = (const float*)d_in[12];
  const float* q2_b = (const float*)d_in[13];
  const float* k2_w = (const float*)d_in[14];
  const float* k2_s = (const float*)d_in[15];
  const float* k2_b = (const float*)d_in[16];
  const float* o_w  = (const float*)d_in[17];
  const float* o_s  = (const float*)d_in[18];
  const float* o_b  = (const float*)d_in[19];
  const float* fuse_w = (const float*)d_in[20];
  const float* fuse_s = (const float*)d_in[21];
  const float* fuse_b = (const float*)d_in[22];
  const float* bott_w = (const float*)d_in[23];
  const float* bott_s = (const float*)d_in[24];
  const float* bott_b = (const float*)d_in[25];

  char* ws = (char*)d_ws;
  auto carve = [&](size_t bytes) -> void* {
    void* p = ws;
    ws += (bytes + 255) & ~(size_t)255;
    return p;
  };
  unsigned short* wq1  = (unsigned short*)carve(2ull * Kk * Tt * Dd);
  unsigned short* wk1  = (unsigned short*)carve(2ull * Kk * Tt * Dd);
  unsigned short* wv_  = (unsigned short*)carve(2ull * Kk * Tt * Dd);
  unsigned short* wq2  = (unsigned short*)carve(2ull * Kk * Tt * Tt);
  unsigned short* wk2  = (unsigned short*)carve(2ull * Kk * Tt * Tt);
  unsigned short* wo   = (unsigned short*)carve(2ull * Kk * CFc * Tt);
  unsigned short* wfu  = (unsigned short*)carve(2ull * CFc * Kk * CFc);
  unsigned short* wbt  = (unsigned short*)carve(2ull * 9 * Oo * CIN);
  unsigned short* wmem = (unsigned short*)carve(2ull * Cc * Kk * CFc);
  unsigned short* Xff  = (unsigned short*)carve(2ull * Bb * CFc);
  unsigned short* Xpm  = (unsigned short*)carve(2ull * Mm * Dd);
  float*          scr  = (float*)carve(4ull * Bb * 64);
  float*          smb  = (float*)carve(4ull * Bb * 64);
  unsigned short* Xk   = (unsigned short*)carve(2ull * Kk * Mm * Dd);
  unsigned short* t1   = (unsigned short*)carve(2ull * Mm * Tt);
  unsigned short* qb   = (unsigned short*)carve(2ull * Mm * Tt);
  unsigned short* kbuf = (unsigned short*)carve(2ull * Mm * Tt);
  unsigned short* vbuf = (unsigned short*)carve(2ull * Mm * Tt);
  unsigned short* Pb   = (unsigned short*)carve(2ull * Bb * 256);
  unsigned short* ctx  = (unsigned short*)carve(2ull * Mm * Tt);
  unsigned short* outk = (unsigned short*)carve(2ull * Mm * CFc);
  float*          facc = (float*)carve(4ull * Mm * CFc);
  unsigned short* xc   = (unsigned short*)carve(2ull * Mm * CIN);

  float* out_mem = (float*)d_out;
  float* out_y   = out_mem + Cc * Kk * CFc;

  const int TPB = 256;
  auto eb = [&](size_t n) { return (unsigned)((n + TPB - 1) / TPB); };   // elementwise blocks
  auto wb = [&](size_t waves) { return (unsigned)((waves + 7) / 8); };   // 8 waves / block

  // weight / memory conversions
  k_cvt<<<eb(Kk * Tt * Dd), TPB, 0, stream>>>(q1_w, wq1, Kk * Tt * Dd);
  k_cvt<<<eb(Kk * Tt * Dd), TPB, 0, stream>>>(k1_w, wk1, Kk * Tt * Dd);
  k_cvt<<<eb(Kk * Tt * Dd), TPB, 0, stream>>>(v_w,  wv_, Kk * Tt * Dd);
  k_cvt<<<eb(Kk * Tt * Tt), TPB, 0, stream>>>(q2_w, wq2, Kk * Tt * Tt);
  k_cvt<<<eb(Kk * Tt * Tt), TPB, 0, stream>>>(k2_w, wk2, Kk * Tt * Tt);
  k_cvt<<<eb(Kk * CFc * Tt), TPB, 0, stream>>>(o_w, wo, Kk * CFc * Tt);
  k_cvt<<<eb((size_t)CFc * Kk * CFc), TPB, 0, stream>>>(fuse_w, wfu, CFc * Kk * CFc);
  k_cvt<<<eb(Cc * Kk * CFc), TPB, 0, stream>>>(memory, wmem, Cc * Kk * CFc);
  k_cvt_bott<<<eb(9 * Oo * CIN), TPB, 0, stream>>>(bott_w, wbt);
  k_prep<<<eb((size_t)Bb * CFc), TPB, 0, stream>>>(feats, Xff, Xpm);

  // memory read: scores GEMM (rows n = c*4+k of flat memory) -> softmax -> sel
  k_gemm4<<<wb((Bb / 64) * (64 / 64)), TPB, 0, stream>>>(
      Xff, CFc, wmem, CFc, scr, nullptr, 64, nullptr, nullptr, Bb, 64, CFc, 0);
  k_smx<<<eb(Bb * Kk), TPB, 0, stream>>>(scr, smb);
  k_sel<<<eb((size_t)Kk * Mm * 64), TPB, 0, stream>>>(smb, memory, Xk);

  for (int k = 0; k < Kk; ++k) {
    const unsigned short* Xkk = Xk + (size_t)k * Mm * Dd;
    // q = cbr(cbr(feats, q1), q2)
    k_gemm4<<<wb((Mm / 64) * (Tt / 64)), TPB, 0, stream>>>(
        Xpm, Dd, wq1 + (size_t)k * Tt * Dd, Dd, nullptr, t1, Tt,
        q1_s + k * Tt, q1_b + k * Tt, Mm, Tt, Dd, 2);
    k_gemm4<<<wb((Mm / 64) * (Tt / 64)), TPB, 0, stream>>>(
        t1, Tt, wq2 + (size_t)k * Tt * Tt, Tt, nullptr, qb, Tt,
        q2_s + k * Tt, q2_b + k * Tt, Mm, Tt, Tt, 2);
    // k = cbr(cbr(kf, k1), k2)
    k_gemm4<<<wb((Mm / 64) * (Tt / 64)), TPB, 0, stream>>>(
        Xkk, Dd, wk1 + (size_t)k * Tt * Dd, Dd, nullptr, t1, Tt,
        k1_s + k * Tt, k1_b + k * Tt, Mm, Tt, Dd, 2);
    k_gemm4<<<wb((Mm / 64) * (Tt / 64)), TPB, 0, stream>>>(
        t1, Tt, wk2 + (size_t)k * Tt * Tt, Tt, nullptr, kbuf, Tt,
        k2_s + k * Tt, k2_b + k * Tt, Mm, Tt, Tt, 2);
    // v = cbr(kf, v)
    k_gemm4<<<wb((Mm / 64) * (Tt / 64)), TPB, 0, stream>>>(
        Xkk, Dd, wv_ + (size_t)k * Tt * Dd, Dd, nullptr, vbuf, Tt,
        v_s + k * Tt, v_b + k * Tt, Mm, Tt, Dd, 2);
    // attention core
    k_sim<<<wb(Bb), TPB, 0, stream>>>(qb, kbuf, Pb);
    k_ctx<<<wb((size_t)Bb * (Tt / 16)), TPB, 0, stream>>>(Pb, vbuf, ctx);
    // out projection
    k_gemm4<<<wb((Mm / 64) * (CFc / 64)), TPB, 0, stream>>>(
        ctx, Tt, wo + (size_t)k * CFc * Tt, Tt, nullptr, outk, CFc,
        o_s + k * CFc, o_b + k * CFc, Mm, CFc, Tt, 2);
    // fuse partial: accumulate W-slice [:, k*1024:(k+1)*1024]
    k_gemm4<<<wb((Mm / 64) * (CFc / 64)), TPB, 0, stream>>>(
        outk, CFc, wfu + (size_t)k * CFc, Kk * CFc, facc, nullptr, CFc,
        nullptr, nullptr, Mm, CFc, CFc, k == 0 ? 0 : 1);
  }

  // fuse epilogue + concat into bott input, then 3x3 conv
  k_fuse_ep<<<eb((size_t)Mm * CFc), TPB, 0, stream>>>(facc, fuse_s, fuse_b, xc);
  k_xcf<<<eb((size_t)Mm * 64), TPB, 0, stream>>>(Xpm, xc);
  k_bott<<<wb((size_t)(Bb / 4) * (Oo / 64)), TPB, 0, stream>>>(xc, wbt, bott_s, bott_b, out_y);

  // first output of the tuple: memory passthrough
  k_copymem<<<eb(Cc * Kk * CFc), TPB, 0, stream>>>(memory, out_mem);
}
